// NodeClassifier_68899865362997
// MI455X (gfx1250) — matrix operand order repaired and checked by reading, not compiled, not run on hardware
//
#include <hip/hip_runtime.h>
#include <hip/hip_bf16.h>

// Problem constants (fixed by the reference)
#define F_IN     128
#define HID      64
#define NCLS     40
#define NCLS_PAD 48

typedef __attribute__((ext_vector_type(16))) __bf16       v16bf;
typedef __attribute__((ext_vector_type(8)))  float        v8f;
typedef __attribute__((ext_vector_type(4)))  unsigned int u32x4;

struct Pack32 { u32x4 lo, hi; };

// Build a 16-element bf16 WMMA fragment from two 16-byte global loads.
static __device__ __forceinline__ v16bf load_frag16(const unsigned short* p0,
                                                    const unsigned short* p1) {
  Pack32 pk;
  pk.lo = *(const u32x4*)p0;
  pk.hi = *(const u32x4*)p1;
  return __builtin_bit_cast(v16bf, pk);
}

// fp32 -> bf16 bits, round-to-nearest-even
static __device__ __forceinline__ unsigned short f2bf(float f) {
  unsigned int u = __builtin_bit_cast(unsigned int, f);
  u += 0x7FFFu + ((u >> 16) & 1u);
  return (unsigned short)(u >> 16);
}

// ---------------------------------------------------------------- utilities
__global__ void k_fill(float* __restrict__ p, float v, long long n) {
  long long i  = (long long)blockIdx.x * blockDim.x + threadIdx.x;
  long long st = (long long)gridDim.x * blockDim.x;
  for (; i < n; i += st) p[i] = v;
}

__global__ void k_f2bf(const float* __restrict__ s, unsigned short* __restrict__ d,
                       long long n) {
  long long i  = (long long)blockIdx.x * blockDim.x + threadIdx.x;
  long long st = (long long)gridDim.x * blockDim.x;
  for (; i < n; i += st) d[i] = f2bf(s[i]);
}

// W2 [HID x NCLS] f32 -> zero-padded [HID x NCLS_PAD] bf16
__global__ void k_w2pad(const float* __restrict__ w, unsigned short* __restrict__ d) {
  int i = blockIdx.x * blockDim.x + threadIdx.x;
  if (i >= HID * NCLS_PAD) return;
  int k = i / NCLS_PAD, c = i % NCLS_PAD;
  d[i] = (c < NCLS) ? f2bf(w[k * NCLS + c]) : (unsigned short)0;
}

// deg[dst] += 1 per edge (deg pre-filled with 1.0 for the self-loop)
__global__ void k_deg_count(const int* __restrict__ dst, float* __restrict__ deg, int E) {
  int e = blockIdx.x * blockDim.x + threadIdx.x;
  if (e < E) atomicAdd(&deg[dst[e]], 1.0f);
}

__global__ void k_rsqrt(const float* __restrict__ deg, float* __restrict__ dinv, int n) {
  int i = blockIdx.x * blockDim.x + threadIdx.x;
  if (i < n) dinv[i] = rsqrtf(deg[i]);
}

// ---------------------------------------------------------------- WMMA GEMM
// D[row][col<NOUT] = dinv[row] * sum_k A[row][k] * B[k][col]
// A: [Nrows x K] bf16 row-major, B: [K x NCOLS] bf16 row-major (NCOLS = padded)
// One wave per 16x16 output tile; blockDim.x = 32 * (NCOLS/16).
template <int K, int NCOLS, int NOUT>
__global__ void k_gemm_bf16(const unsigned short* __restrict__ A,
                            const unsigned short* __restrict__ B,
                            const float* __restrict__ dinv,
                            float* __restrict__ D, int Nrows) {
  const int lane = threadIdx.x & 31;
  const int wave = threadIdx.x >> 5;
  const int m0   = blockIdx.x << 4;
  const int n0   = wave << 4;
  const int half = lane >> 4;   // 0: lanes 0-15, 1: lanes 16-31
  const int l    = lane & 15;

  int row = m0 + l;
  if (row >= Nrows) row = Nrows - 1;   // clamp: keep EXEC all-1s for WMMA

  v8f acc = {};
#pragma unroll
  for (int k0 = 0; k0 < K; k0 += 32) {
    // A 16x32 bf16 fragment: lane<16 holds K {0..7,16..23}, lane>=16 {8..15,24..31}
    const unsigned short* pa = A + (size_t)row * K + (k0 + half * 8);
    v16bf a = load_frag16(pa, pa + 16);
    // B 32x16 bf16 fragment: lane holds column N = n0+l, K {0..7,16..23} (lane<16) / {8..15,24..31} (lane>=16)
    const unsigned short* pb = B + (size_t)(k0 + half * 8) * NCOLS + n0 + l;
    v16bf b;
#pragma unroll
    for (int s = 0; s < 16; ++s)
      b[s] = __builtin_bit_cast(__bf16, pb[(size_t)(s < 8 ? s : s + 8) * NCOLS]);
    acc = __builtin_amdgcn_wmma_f32_16x16x32_bf16(false, a, false, b,
                                                  (short)0, acc, false, false);
  }

  // C/D layout: VGPR r, lanes 0-15 -> M=r, lanes 16-31 -> M=8+r; N = lane&15
  const int  rbase    = m0 + half * 8;
  const int  col      = n0 + l;
  const bool colOk    = (col < NOUT);
  float*     drow     = D + (size_t)rbase * NOUT + col;

  if (m0 + 16 <= Nrows) {            // uniform (scalar) branch: full tile, fast path
    if (colOk) {
      float dv[8];
#pragma unroll
      for (int r = 0; r < 8; ++r) dv[r] = dinv[rbase + r];
#pragma unroll
      for (int r = 0; r < 8; ++r) drow[(size_t)r * NOUT] = acc[r] * dv[r];
    }
  } else {                           // ragged last tile (not hit when Nrows % 16 == 0)
    if (colOk) {
#pragma unroll
      for (int r = 0; r < 8; ++r) {
        const int rr = rbase + r;
        if (rr < Nrows) D[(size_t)rr * NOUT + col] = acc[r] * dinv[rr];
      }
    }
  }
}

// ------------------------------------------------------- edge scatter-add
// One wave per edge: agg[dst] += hs[src] over F features (coalesced 128B segs)
__global__ void k_agg(const int* __restrict__ src, const int* __restrict__ dst,
                      const float* __restrict__ hs, float* __restrict__ agg,
                      int E, int F) {
  int e = (int)(((long long)blockIdx.x * blockDim.x + threadIdx.x) >> 5);
  if (e >= E) return;
  int lane = threadIdx.x & 31;
  int s = src[e], d = dst[e];
  const float* hrow = hs  + (size_t)s * F;
  float*       arow = agg + (size_t)d * F;
  for (int f = lane; f < F; f += 32) atomicAdd(&arow[f], hrow[f]);
}

// -------------------------------------------------------------- epilogues
// layer1: t = dinv[i]*(agg + self) + b1; relu; emit bf16 for layer-2 GEMM
__global__ void k_post1(const float* __restrict__ agg, const float* __restrict__ hs,
                        const float* __restrict__ dinv, const float* __restrict__ b1,
                        unsigned short* __restrict__ h2, long long total) {
  long long idx = (long long)blockIdx.x * blockDim.x + threadIdx.x;
  if (idx >= total) return;
  int i = (int)(idx >> 6);           // /HID (64)
  int f = (int)(idx & 63);
  float t = dinv[i] * (agg[idx] + hs[idx]) + b1[f];
  h2[idx] = f2bf(t > 0.0f ? t : 0.0f);
}

// layer2: out = dinv[i]*(agg + self) + b2
__global__ void k_post2(const float* __restrict__ agg, const float* __restrict__ zs,
                        const float* __restrict__ dinv, const float* __restrict__ b2,
                        float* __restrict__ out, long long total) {
  long long idx = (long long)blockIdx.x * blockDim.x + threadIdx.x;
  if (idx >= total) return;
  int i = (int)(idx / NCLS);
  int c = (int)(idx % NCLS);
  out[idx] = dinv[i] * (agg[idx] + zs[idx]) + b2[c];
}

// ---------------------------------------------------------------- launcher
extern "C" void kernel_launch(void* const* d_in, const int* in_sizes, int n_in,
                              void* d_out, int out_size, void* d_ws, size_t ws_size,
                              hipStream_t stream) {
  const float* x  = (const float*)d_in[0];
  const int*   ei = (const int*)d_in[1];
  const float* W1 = (const float*)d_in[2];
  const float* b1 = (const float*)d_in[3];
  const float* W2 = (const float*)d_in[4];
  const float* b2 = (const float*)d_in[5];
  float* out = (float*)d_out;

  const int Nn = in_sizes[0] / F_IN;   // 100000
  const int E  = in_sizes[1] / 2;      // 1600000
  const int* src = ei;
  const int* dst = ei + E;

  // Workspace carve-out (256B aligned)
  char*  w   = (char*)d_ws;
  size_t off = 0;
  auto alloc = [&](size_t bytes) -> char* {
    char* p = w + off;
    off = (off + bytes + 255) & ~(size_t)255;
    return p;
  };
  float*          deg  = (float*)alloc((size_t)Nn * 4);
  float*          dinv = (float*)alloc((size_t)Nn * 4);
  unsigned short* xb   = (unsigned short*)alloc((size_t)Nn * F_IN * 2);
  unsigned short* w1b  = (unsigned short*)alloc((size_t)F_IN * HID * 2);
  unsigned short* w2b  = (unsigned short*)alloc((size_t)HID * NCLS_PAD * 2);
  float*          hs   = (float*)alloc((size_t)Nn * HID * 4);  // layer1 h*dinv; reused as z*dinv
  float*          agg  = (float*)alloc((size_t)Nn * HID * 4);  // layer1 agg; reused layer2
  unsigned short* h2   = (unsigned short*)alloc((size_t)Nn * HID * 2);

  const int TB = 256;
  const int Mtiles = (Nn + 15) / 16;

  // ---- degree / normalization
  k_fill<<<2048, TB, 0, stream>>>(deg, 1.0f, (long long)Nn);             // self-loop
  k_deg_count<<<(E + TB - 1) / TB, TB, 0, stream>>>(dst, deg, E);
  k_rsqrt<<<(Nn + TB - 1) / TB, TB, 0, stream>>>(deg, dinv, Nn);

  // ---- bf16 conversions
  k_f2bf<<<2048, TB, 0, stream>>>(x,  xb,  (long long)Nn * F_IN);
  k_f2bf<<<(F_IN * HID + TB - 1) / TB, TB, 0, stream>>>(W1, w1b, (long long)F_IN * HID);
  k_w2pad<<<(HID * NCLS_PAD + TB - 1) / TB, TB, 0, stream>>>(W2, w2b);

  // ---- layer 1: GEMM (pre-scaled) -> aggregate -> post (+b1, ReLU, ->bf16)
  k_gemm_bf16<F_IN, HID, HID><<<Mtiles, 32 * (HID / 16), 0, stream>>>(xb, w1b, dinv, hs, Nn);
  k_fill<<<2048, TB, 0, stream>>>(agg, 0.0f, (long long)Nn * HID);
  {
    long long thr = (long long)E * 32;
    k_agg<<<(unsigned)((thr + TB - 1) / TB), TB, 0, stream>>>(src, dst, hs, agg, E, HID);
  }
  {
    long long tot = (long long)Nn * HID;
    k_post1<<<(unsigned)((tot + TB - 1) / TB), TB, 0, stream>>>(agg, hs, dinv, b1, h2, tot);
  }

  // ---- layer 2: GEMM (pre-scaled) -> aggregate -> post (+b2)
  k_fill<<<2048, TB, 0, stream>>>(agg, 0.0f, (long long)Nn * NCLS);
  k_gemm_bf16<HID, NCLS_PAD, NCLS><<<Mtiles, 32 * (NCLS_PAD / 16), 0, stream>>>(h2, w2b, dinv, hs, Nn);
  {
    long long thr = (long long)E * 32;
    k_agg<<<(unsigned)((thr + TB - 1) / TB), TB, 0, stream>>>(src, dst, hs, agg, E, NCLS);
  }
  {
    long long tot = (long long)Nn * NCLS;
    k_post2<<<(unsigned)((tot + TB - 1) / TB), TB, 0, stream>>>(agg, hs, dinv, b2, out, tot);
  }
}
